// Encoder_55585466745419
// MI455X (gfx1250) — matrix operand order repaired and checked by reading, not compiled, not run on hardware
//
#include <hip/hip_runtime.h>
#include <hip/hip_bf16.h>

typedef __attribute__((ext_vector_type(16))) _Float16 v16h;
typedef __attribute__((ext_vector_type(4)))  _Float16 v4h;
typedef __attribute__((ext_vector_type(8)))  float    v8f;
typedef __attribute__((ext_vector_type(4)))  float    v4f;

#define BK 32

// ---------------------------------------------------------------------------
// Batched GEMM:  C = act( scale * (A @ B) + bias )
//   A: M x K (row-major, lda)                  fp32 -> f16 LDS [m][k]
//   B: K x N (row-major, ldb)  if BT==0        fp32 -> f16 LDS [n][k]
//      N x K (row-major, ldb)  if BT==1
//   C: M x N (row-major, ldc)  fp32
// blockIdx.z = bb*Hh + hb with independent (bb,hb) pointer strides.
// Software-pipelined: global->reg staging for tile k+1 overlaps WMMA on tile k;
// double-buffered LDS -> one barrier per K-step.
// ---------------------------------------------------------------------------
template<int TBM, int TBN, int TWM, int TWN, int BT>
__global__ __launch_bounds__((TBM / TWM) * (TBN / TWN) * 32) void gemm_wmma(
    const float* __restrict__ A, const float* __restrict__ Bm,
    const float* __restrict__ bias, float* __restrict__ C,
    int Kd, int lda, int ldb, int ldc,
    float scale, int relu, int Hh,
    long long sAb, long long sAh, long long sBb, long long sBh,
    long long sCb, long long sCh)
{
    constexpr int NWAVE = (TBM / TWM) * (TBN / TWN);
    constexpr int NT    = NWAVE * 32;
    constexpr int AFR   = TWM / 16;              // A fragments per wave
    constexpr int BFR   = TWN / 16;              // B fragments per wave
    constexpr int PA    = (TBM * BK) / (4 * NT); // float4 staging passes (A)
    constexpr int PB    = (TBN * BK) / (4 * NT); // float4 staging passes (B)
    constexpr int RSTEP = NT / 8;                // rows covered per pass ([.][k] layout)
    constexpr int KSTEP = NT / (TBN / 4);        // k-rows per pass (non-trans B)

    __shared__ _Float16 As[2][TBM][BK];          // [buf][m][k]
    __shared__ _Float16 Bs[2][TBN][BK];          // [buf][n][k]

    const int bz = blockIdx.z;
    const int bb = bz / Hh, hb = bz % Hh;
    A  += (long long)bb * sAb + (long long)hb * sAh;
    Bm += (long long)bb * sBb + (long long)hb * sBh;
    C  += (long long)bb * sCb + (long long)hb * sCh;

    const int m0 = blockIdx.y * TBM;
    const int n0 = blockIdx.x * TBN;
    const int tid  = threadIdx.x;
    const int wave = tid >> 5;
    const int lane = tid & 31;
    const int wm = (wave / (TBN / TWN)) * TWM;
    const int wn = (wave % (TBN / TWN)) * TWN;
    const int rl = lane & 15;
    const int hs = lane >> 4;                    // lane-half select

    // per-thread staging coordinates
    const int ar  = tid >> 3, acg = tid & 7;                 // [row][float4-group] for A / BT-B
    const int bkk = tid / (TBN / 4), bng = tid % (TBN / 4);  // non-trans B

    const float* Apt  = A  + (long long)(m0 + ar) * lda + acg * 4;
    const float* Bpt  = Bm + (long long)(n0 + ar) * ldb + acg * 4;        // BT==1
    const float* Bpn  = Bm + (long long)bkk * ldb + n0 + bng * 4;         // BT==0

    v4f ta[PA], tb[PB];

    auto gload = [&](int k0) {
        #pragma unroll
        for (int p = 0; p < PA; ++p)
            ta[p] = *(const v4f*)(Apt + (long long)(p * RSTEP) * lda + k0);
        if (BT) {
            #pragma unroll
            for (int p = 0; p < PB; ++p)
                tb[p] = *(const v4f*)(Bpt + (long long)(p * RSTEP) * ldb + k0);
        } else {
            #pragma unroll
            for (int p = 0; p < PB; ++p)
                tb[p] = *(const v4f*)(Bpn + (long long)(k0 + p * KSTEP) * ldb);
        }
    };

    auto sstore = [&](int buf) {
        #pragma unroll
        for (int p = 0; p < PA; ++p) {
            v4h h4;
            #pragma unroll
            for (int j = 0; j < 4; ++j) h4[j] = (_Float16)ta[p][j];
            *(v4h*)&As[buf][ar + p * RSTEP][acg * 4] = h4;
        }
        if (BT) {
            #pragma unroll
            for (int p = 0; p < PB; ++p) {
                v4h h4;
                #pragma unroll
                for (int j = 0; j < 4; ++j) h4[j] = (_Float16)tb[p][j];
                *(v4h*)&Bs[buf][ar + p * RSTEP][acg * 4] = h4;
            }
        } else {
            #pragma unroll
            for (int p = 0; p < PB; ++p)
                #pragma unroll
                for (int j = 0; j < 4; ++j)
                    Bs[buf][bng * 4 + j][bkk + p * KSTEP] = (_Float16)tb[p][j];
        }
    };

    v8f acc[AFR][BFR] = {};
    const int nk = Kd / BK;

    gload(0);
    sstore(0);
    __syncthreads();

    for (int kb = 0; kb < nk; ++kb) {
        const int cur = kb & 1, nxt = cur ^ 1;

        // issue next tile's global loads early (overlap with WMMA below)
        if (kb + 1 < nk) gload((kb + 1) * BK);
        // prefetch tile k+2 (global_prefetch_b8)
        if (kb + 2 < nk) {
            __builtin_prefetch(Apt + (long long)(kb + 2) * BK, 0, 3);
            if (BT) __builtin_prefetch(Bpt + (long long)(kb + 2) * BK, 0, 3);
            else    __builtin_prefetch(Bpn + (long long)((kb + 2) * BK) * ldb, 0, 3);
        }

        // ---- build fragments (CDNA5 16-bit WMMA layouts) + WMMA ----
        v16h af[AFR], bf[BFR];
        #pragma unroll
        for (int i = 0; i < AFR; ++i) {
            const _Float16* pa = &As[cur][wm + i * 16 + rl][0];
            #pragma unroll
            for (int e = 0; e < 8; ++e) af[i][e]     = pa[16 * hs + e];      // K {0..7}/{16..23}
            #pragma unroll
            for (int e = 0; e < 8; ++e) af[i][8 + e] = pa[16 * hs + 8 + e];  // K {8..15}/{24..31}
        }
        #pragma unroll
        for (int j = 0; j < BFR; ++j) {
            const _Float16* pb = &Bs[cur][wn + j * 16 + rl][0];
            #pragma unroll
            for (int e = 0; e < 16; ++e) bf[j][e] = pb[16 * hs + e];         // K {0..15}/{16..31}
        }
        #pragma unroll
        for (int i = 0; i < AFR; ++i)
            #pragma unroll
            for (int j = 0; j < BFR; ++j)
                acc[i][j] = __builtin_amdgcn_wmma_f32_16x16x32_f16(
                    false, af[i], false, bf[j], (short)0, acc[i][j], false, false);

        // convert + store next tile into the other LDS buffer
        if (kb + 1 < nk) sstore(nxt);
        __syncthreads();
    }

    // ---- store: f32 C/D layout: N = lane%16, M = vgpr + 8*(lane>=16) ----
    const int cn = lane & 15, ch = lane >> 4;
    #pragma unroll
    for (int i = 0; i < AFR; ++i)
        #pragma unroll
        for (int j = 0; j < BFR; ++j) {
            int gn = n0 + wn + j * 16 + cn;
            float bvv = bias ? bias[gn] : 0.0f;
            #pragma unroll
            for (int r = 0; r < 8; ++r) {
                int gm = m0 + wm + i * 16 + (ch << 3) + r;
                float v = acc[i][j][r] * scale + bvv;
                if (relu) v = fmaxf(v, 0.0f);
                C[(long long)gm * ldc + gn] = v;
            }
        }
}

// ---------------------------------------------------------------------------
// LayerNorm over D=1024: Y = LN(X + R) * g + b   (R optional). 1 block / row.
// ---------------------------------------------------------------------------
__global__ __launch_bounds__(256) void ln1024(
    const float* __restrict__ X, const float* __restrict__ R,
    const float* __restrict__ g, const float* __restrict__ bta,
    float* __restrict__ Y)
{
    const long long row = blockIdx.x;
    const float* x = X + row * 1024;
    const float* r = R ? R + row * 1024 : nullptr;
    float* y = Y + row * 1024;
    __shared__ float sd[256];
    const int t = threadIdx.x;

    float v[4];
    float s = 0.0f;
    #pragma unroll
    for (int i = 0; i < 4; ++i) {
        int c = t + i * 256;
        float z = x[c];
        if (r) z += r[c];
        v[i] = z; s += z;
    }
    sd[t] = s; __syncthreads();
    for (int st = 128; st > 0; st >>= 1) { if (t < st) sd[t] += sd[t + st]; __syncthreads(); }
    float mu = sd[0] * (1.0f / 1024.0f);
    __syncthreads();

    float ss = 0.0f;
    #pragma unroll
    for (int i = 0; i < 4; ++i) { float d = v[i] - mu; ss += d * d; }
    sd[t] = ss; __syncthreads();
    for (int st = 128; st > 0; st >>= 1) { if (t < st) sd[t] += sd[t + st]; __syncthreads(); }
    float rstd = rsqrtf(sd[0] * (1.0f / 1024.0f) + 1e-5f);

    #pragma unroll
    for (int i = 0; i < 4; ++i) {
        int c = t + i * 256;
        y[c] = (v[i] - mu) * rstd * g[c] + bta[c];
    }
}

// ---------------------------------------------------------------------------
// Masked softmax over rows of length 512 (in place). 1 block / row.
// ---------------------------------------------------------------------------
__global__ __launch_bounds__(256) void softmax512(
    float* __restrict__ attn, const int* __restrict__ mask)
{
    const long long rid = blockIdx.x;
    float* p = attn + rid * 512;
    const int b = (int)(rid / (16LL * 512LL));
    const int* mrow = mask + (long long)b * 512;
    __shared__ float sd[256];
    const int t = threadIdx.x;

    float v0 = p[t], v1 = p[t + 256];
    if (mrow[t] == 0)       v0 = -1e10f;
    if (mrow[t + 256] == 0) v1 = -1e10f;

    sd[t] = fmaxf(v0, v1); __syncthreads();
    for (int st = 128; st > 0; st >>= 1) { if (t < st) sd[t] = fmaxf(sd[t], sd[t + st]); __syncthreads(); }
    float mx = sd[0]; __syncthreads();

    float e0 = __expf(v0 - mx), e1 = __expf(v1 - mx);
    sd[t] = e0 + e1; __syncthreads();
    for (int st = 128; st > 0; st >>= 1) { if (t < st) sd[t] += sd[t + st]; __syncthreads(); }
    float inv = 1.0f / sd[0];

    p[t] = e0 * inv;
    p[t + 256] = e1 * inv;
}

// ---------------------------------------------------------------------------
// Host side
// ---------------------------------------------------------------------------
extern "C" void kernel_launch(void* const* d_in, const int* in_sizes, int n_in,
                              void* d_out, int out_size, void* d_ws, size_t ws_size,
                              hipStream_t stream) {
    (void)in_sizes; (void)n_in; (void)out_size; (void)ws_size;

    const int Bn = 8, Sn = 512, Dn = 1024, Hn = 16, DFn = 4096;
    const int MR = Bn * Sn;                               // 4096 token rows
    const long long SZ_D  = (long long)MR * Dn;           // 4,194,304
    const long long SZ_F  = (long long)MR * DFn;          // 16,777,216
    const long long SZ_AL = (long long)Bn * Hn * Sn * Sn; // 33,554,432 / layer

    const float* src_x  = (const float*)d_in[0];
    const float* src_m  = (const float*)d_in[1];
    const int*   mask   = (const int*)  d_in[2];
    const float* ln1xg  = (const float*)d_in[3];
    const float* ln1xb  = (const float*)d_in[4];
    const float* ln1mg  = (const float*)d_in[5];
    const float* ln1mb  = (const float*)d_in[6];
    const float* Wq     = (const float*)d_in[7];
    const float* bq     = (const float*)d_in[8];
    const float* Wk     = (const float*)d_in[9];
    const float* bk     = (const float*)d_in[10];
    const float* Wv     = (const float*)d_in[11];
    const float* bv     = (const float*)d_in[12];
    const float* Wxo    = (const float*)d_in[13];
    const float* bxo    = (const float*)d_in[14];
    const float* Wmo    = (const float*)d_in[15];
    const float* bmo    = (const float*)d_in[16];
    const float* ln2xg  = (const float*)d_in[17];
    const float* ln2xb  = (const float*)d_in[18];
    const float* ln2mg  = (const float*)d_in[19];
    const float* ln2mb  = (const float*)d_in[20];
    const float* Wx1    = (const float*)d_in[21];
    const float* bx1    = (const float*)d_in[22];
    const float* Wx2    = (const float*)d_in[23];
    const float* bx2    = (const float*)d_in[24];
    const float* Wm1    = (const float*)d_in[25];
    const float* bm1    = (const float*)d_in[26];
    const float* Wm2    = (const float*)d_in[27];
    const float* bm2    = (const float*)d_in[28];
    const float* ln3xg  = (const float*)d_in[29];
    const float* ln3xb  = (const float*)d_in[30];
    const float* ln3mg  = (const float*)d_in[31];
    const float* ln3mb  = (const float*)d_in[32];

    float* w   = (float*)d_ws;
    float* sx  = w;                 w += SZ_D;
    float* sm  = w;                 w += SZ_D;
    float* xn  = w;                 w += SZ_D;
    float* mn  = w;                 w += SZ_D;
    float* Qb  = w;                 w += SZ_D;
    float* Kb  = w;                 w += SZ_D;
    float* Vb  = w;                 w += SZ_D;
    float* kbb = w;                 w += SZ_D;
    float* vbb = w;                 w += SZ_D;
    float* xa  = w;                 w += SZ_D;
    float* ma  = w;                 w += SZ_D;
    float* sx2 = w;                 w += SZ_D;
    float* sm2 = w;                 w += SZ_D;
    float* fx  = w;                 w += SZ_D;
    float* fm  = w;                 w += SZ_D;
    float* h1  = w;                 w += SZ_F;

    float* out      = (float*)d_out;
    float* attn_out = out + 2 * SZ_D;

    hipMemcpyAsync(sx, src_x, SZ_D * sizeof(float), hipMemcpyDeviceToDevice, stream);
    hipMemcpyAsync(sm, src_m, SZ_D * sizeof(float), hipMemcpyDeviceToDevice, stream);

    const long long sBS = (long long)Sn * Dn;   // 524288: per-batch token-mat stride
    const long long sSS = (long long)Sn * Sn;   // 262144

    // dense: 128x128 block tile, 8 waves x (64x32) each
    auto dense = [&](const float* A, const float* B, const float* bias, float* C,
                     int M, int N, int K, int lda, int ldb, int ldc, int relu) {
        gemm_wmma<128, 128, 64, 32, 0><<<dim3(N / 128, M / 128, 1), 256, 0, stream>>>(
            A, B, bias, C, K, lda, ldb, ldc, 1.0f, relu, 1, 0, 0, 0, 0, 0, 0);
    };

    for (int l = 0; l < 6; ++l) {
        const long long oD  = (long long)l * Dn;
        const long long oDD = (long long)l * Dn * Dn;
        const long long oDF = (long long)l * Dn * DFn;
        const long long oF  = (long long)l * DFn;
        float* attn_l = attn_out + (long long)l * SZ_AL;

        // 1) pre-LN of both streams
        ln1024<<<MR, 256, 0, stream>>>(sx, nullptr, ln1xg + oD, ln1xb + oD, xn);
        ln1024<<<MR, 256, 0, stream>>>(sm, nullptr, ln1mg + oD, ln1mb + oD, mn);

        // 2) Q,K from x-stream; V from m-stream
        dense(xn, Wq + oDD, bq + oD, Qb, MR, Dn, Dn, Dn, Dn, Dn, 0);
        dense(xn, Wk + oDD, bk + oD, Kb, MR, Dn, Dn, Dn, Dn, Dn, 0);
        dense(mn, Wv + oDD, bv + oD, Vb, MR, Dn, Dn, Dn, Dn, Dn, 0);

        // 3) scores = Q K^T / 8 per (b,h): A=[S,64] lda=D; B = K as NxK (BT=1)
        gemm_wmma<128, 128, 64, 32, 1><<<dim3(Sn / 128, Sn / 128, Bn * Hn), 256, 0, stream>>>(
            Qb, Kb, nullptr, attn_l, 64, Dn, Dn, Sn, 0.125f, 0, Hn,
            sBS, 64, sBS, 64, (long long)Hn * sSS, sSS);

        // 4) masked softmax in place (output buffer)
        softmax512<<<Bn * Hn * Sn, 256, 0, stream>>>(attn_l, mask);

        // 5) k_bar = attn @ K, v_bar = attn @ V  (per head, N = DK = 64)
        gemm_wmma<64, 64, 32, 32, 0><<<dim3(1, Sn / 64, Bn * Hn), 128, 0, stream>>>(
            attn_l, Kb, nullptr, kbb, Sn, Sn, Dn, Dn, 1.0f, 0, Hn,
            (long long)Hn * sSS, sSS, sBS, 64, sBS, 64);
        gemm_wmma<64, 64, 32, 32, 0><<<dim3(1, Sn / 64, Bn * Hn), 128, 0, stream>>>(
            attn_l, Vb, nullptr, vbb, Sn, Sn, Dn, Dn, 1.0f, 0, Hn,
            (long long)Hn * sSS, sSS, sBS, 64, sBS, 64);

        // 6) output projections
        dense(kbb, Wxo + oDD, bxo + oD, xa, MR, Dn, Dn, Dn, Dn, Dn, 0);
        dense(vbb, Wmo + oDD, bmo + oD, ma, MR, Dn, Dn, Dn, Dn, Dn, 0);

        // 7) sx2 = LN(sx + xa), sm2 = LN(sm + ma)
        ln1024<<<MR, 256, 0, stream>>>(sx, xa, ln2xg + oD, ln2xb + oD, sx2);
        ln1024<<<MR, 256, 0, stream>>>(sm, ma, ln2mg + oD, ln2mb + oD, sm2);

        // 8) FFN(x): fx = relu(xa@Wx1+b)@Wx2+b ; sx = LN(sx2 + fx)
        dense(xa, Wx1 + oDF, bx1 + oF, h1, MR, DFn, Dn, Dn, DFn, DFn, 1);
        dense(h1, Wx2 + oDF, bx2 + oD, fx, MR, Dn, DFn, DFn, Dn, Dn, 0);
        ln1024<<<MR, 256, 0, stream>>>(sx2, fx, ln3xg + oD, ln3xb + oD, sx);

        //    FFN(m): fm = relu(ma@Wm1+b)@Wm2+b ; sm = LN(sm2 + fm)
        dense(ma, Wm1 + oDF, bm1 + oF, h1, MR, DFn, Dn, Dn, DFn, DFn, 1);
        dense(h1, Wm2 + oDF, bm2 + oD, fm, MR, Dn, DFn, DFn, Dn, Dn, 0);
        ln1024<<<MR, 256, 0, stream>>>(sm2, fm, ln3mg + oD, ln3mb + oD, sm);
    }

    hipMemcpyAsync(out,        sx, SZ_D * sizeof(float), hipMemcpyDeviceToDevice, stream);
    hipMemcpyAsync(out + SZ_D, sm, SZ_D * sizeof(float), hipMemcpyDeviceToDevice, stream);
}
